// ViT_52226802320101
// MI455X (gfx1250) — matrix operand order repaired and checked by reading, not compiled, not run on hardware
//
#include <hip/hip_runtime.h>
#include <hip/hip_bf16.h>
#include <hip/hip_fp16.h>

typedef __attribute__((ext_vector_type(16))) __bf16         v16bf;
typedef __attribute__((ext_vector_type(16))) unsigned short v16us;
typedef __attribute__((ext_vector_type(8)))  float          v8f;
typedef __attribute__((ext_vector_type(4)))  unsigned int   v4u;
typedef __attribute__((ext_vector_type(8)))  int            v8i;
typedef __attribute__((ext_vector_type(4)))  int            v4i;

#define BSZ     16
#define CCH     3
#define HIMG    224
#define PSZ     16
#define GRD     14
#define NTOK    196
#define DMODEL  768
#define NHEAD   12
#define DHEAD   64
#define FDIM    3072
#define NLAYER  12
#define MROWS   (BSZ * NTOK)   /* 3136 */
#define NPAD    208
#define EPSLN   1e-5f

#define ASTR    48   /* LDS stride (halves) for A planes, 96B = 16B aligned */
#define BSTR    48   /* LDS stride (halves) for transposed B planes */

#if defined(__has_builtin)
#if __has_builtin(__builtin_amdgcn_tensor_load_to_lds)
#define HAVE_TDM 1
#endif
#endif
#ifndef HAVE_TDM
#define HAVE_TDM 0
#endif

// ---------------------------------------------------------------------------
// WMMA helpers (CDNA5 wave32, D = A(16x32 bf16) x B(32x16 bf16) + C(16x16 f32))
// ---------------------------------------------------------------------------
__device__ __forceinline__ v8f wmma_bf16(v16bf a, v16bf b, v8f c) {
  // 8 args: (neg_a, A, neg_b, B, c_mod, C, reuse_a, reuse_b)
  return __builtin_amdgcn_wmma_f32_16x16x32_bf16(false, a, false, b, (short)0, c,
                                                 false, false);
}

// A fragment: 16x32 bf16, lanes 0-15: M=m, K in {0..15}; lanes 16-31: K in {16..31}
__device__ __forceinline__ v16bf fragA(const __bf16* plane, int mbase, int lane) {
  const __bf16* rowp = plane + (mbase + (lane & 15)) * ASTR;
  int kadd = (lane >> 4) ? 16 : 0;
  v16bf r;
#pragma unroll
  for (int h = 0; h < 16; ++h) {
    int k = h + kadd;
    r[h] = rowp[k];
  }
  return r;
}

// B fragment from LDS stored transposed as BT[col][k]:
// lane = col (0-15 / 16-31), K = h + 16*(lane>=16)  (16 contiguous halves)
__device__ __forceinline__ v16bf fragB(const __bf16* plane, int cbase, int lane) {
  const __bf16* rowp = plane + (cbase + (lane & 15)) * BSTR + ((lane >> 4) << 4);
  v16bf r;
#pragma unroll
  for (int h = 0; h < 16; ++h) r[h] = rowp[h];
  return r;
}

#if HAVE_TDM
// ---------------------------------------------------------------------------
// TDM: DMA a 2D tile of float2 (8B elems), tile_dim0=64 cols x tile_dim1=32
// rows, row stride = n_stride elems, from global into LDS (row-major, packed).
// Issued by one lane of wave 0; EXEC is ignored by TENSOR_* ops.
// ---------------------------------------------------------------------------
__device__ __forceinline__ void tdm_load_tile_f2(const float2* gsrc,
                                                 unsigned lds_byte_off,
                                                 int n_stride, int rows_avail) {
  unsigned long long ga = (unsigned long long)(uintptr_t)gsrc;
  v4u g0;
  g0[0] = 1u;                                   // count=1, user descriptor
  g0[1] = lds_byte_off;                          // lds_addr
  g0[2] = (unsigned)(ga & 0xFFFFFFFFull);        // global_addr[31:0]
  g0[3] = (unsigned)((ga >> 32) & 0x1FFFFFFull)  // global_addr[56:32]
          | (2u << 30);                          // type = 2 ("image")
  unsigned td0 = (unsigned)n_stride;             // tensor_dim0 (elements)
  unsigned td1 = (unsigned)rows_avail;           // tensor_dim1 (rows, OOB fence)
  v8i g1;
  g1[0] = (int)(3u << 16);                       // data_size=3 (8B), no mcast
  g1[1] = (int)((td0 & 0xFFFFu) << 16);          // dim0[15:0] @ bits 63:48
  g1[2] = (int)((td0 >> 16) | ((td1 & 0xFFFFu) << 16));
  g1[3] = (int)((td1 >> 16) | (64u << 16));      // tile_dim0 = 64
  g1[4] = (int)32u;                              // tile_dim1 = 32, tile_dim2 = 0
  g1[5] = (int)td0;                              // tensor_dim0_stride lo32
  g1[6] = 0;
  g1[7] = 0;
  v4i gz = {0, 0, 0, 0};
#if defined(__clang_major__) && (__clang_major__ >= 23)
  v8i gz8 = {0, 0, 0, 0, 0, 0, 0, 0};
  __builtin_amdgcn_tensor_load_to_lds(g0, g1, gz, gz, gz8, 0);
#else
  __builtin_amdgcn_tensor_load_to_lds(g0, g1, gz, gz, 0);
#endif
}
#endif

// ---------------------------------------------------------------------------
// Patch packing: x [B,C,224,224] cplx -> A [3136, 768] cplx (row=b*196+gy*14+gx)
// ---------------------------------------------------------------------------
__global__ __launch_bounds__(256) void pack_patches_k(const float2* __restrict__ x,
                                                      float2* __restrict__ A) {
  int idx = blockIdx.x * 256 + threadIdx.x;
  if (idx >= MROWS * DMODEL) return;
  int row = idx / DMODEL, col = idx % DMODEL;
  int b = row / NTOK, t = row % NTOK;
  int gy = t / GRD, gx = t % GRD;
  int c = col >> 8, rem = col & 255;
  int py = rem >> 4, px = rem & 15;
  size_t src = ((size_t)(b * CCH + c) * HIMG + (gy * PSZ + py)) * HIMG + (gx * PSZ + px);
  A[idx] = x[src];
}

// ---------------------------------------------------------------------------
// Generic complex GEMM: C[M,N] = A[M,K] x W[K,N] + bias (+ resid), bf16 WMMA
// Workgroup tile 128x64, 8 waves (4x2), wave tile 32x32 (2x2 of 16x16)
// Cre = (Are.Bre) - (Aim.Bim)  [two accumulators, subtract at epilogue]
// Cim = (Are.Bim) + (Aim.Bre)
// W tile is DMA'd to LDS by the Tensor Data Mover when available.
// ---------------------------------------------------------------------------
__global__ __launch_bounds__(256) void cgemm_cc_k(const float2* __restrict__ A,
                                                  const float2* __restrict__ W,
                                                  const float2* __restrict__ bias,
                                                  const float2* __restrict__ resid,
                                                  float2* __restrict__ C,
                                                  int M, int N, int K) {
  __shared__ __bf16 sAre[128 * ASTR];
  __shared__ __bf16 sAim[128 * ASTR];
  __shared__ __bf16 sBre[64 * BSTR];
  __shared__ __bf16 sBim[64 * BSTR];
#if HAVE_TDM
  __shared__ float2 sWstage[32 * 64];   // fp32 complex tile landed by TDM
#endif
  int tid = threadIdx.x, lane = tid & 31, w = tid >> 5;
  int wm = w >> 1, wn = w & 1;
  int m0 = blockIdx.y * 128, n0 = blockIdx.x * 64;
  const v8f vz = {0.f, 0.f, 0.f, 0.f, 0.f, 0.f, 0.f, 0.f};
  v8f accpp[2][2], accmm[2][2], accim[2][2];
#pragma unroll
  for (int i = 0; i < 2; ++i)
#pragma unroll
    for (int j = 0; j < 2; ++j) { accpp[i][j] = vz; accmm[i][j] = vz; accim[i][j] = vz; }

  for (int k0 = 0; k0 < K; k0 += 32) {
#if HAVE_TDM
    if (tid == 0) {
      tdm_load_tile_f2(W + (size_t)k0 * N + n0,
                       (unsigned)(uintptr_t)(&sWstage[0]), N, K - k0);
    }
#endif
    {  // stage A tile 128x32 (complex fp32 -> bf16 planes)
      int row = tid >> 1, half = tid & 1;
      int m = m0 + row;
      __bf16* dre = &sAre[row * ASTR + half * 16];
      __bf16* di  = &sAim[row * ASTR + half * 16];
      if (m < M) {
        const float2* src = A + (size_t)m * K + k0 + half * 16;
#pragma unroll
        for (int i = 0; i < 16; ++i) {
          float2 v = src[i];
          dre[i] = (__bf16)v.x;
          di[i]  = (__bf16)v.y;
        }
      } else {
#pragma unroll
        for (int i = 0; i < 16; ++i) { dre[i] = (__bf16)0.f; di[i] = (__bf16)0.f; }
      }
    }
#if HAVE_TDM
    if (tid < 32) __builtin_amdgcn_s_wait_tensorcnt((short)0);
    __syncthreads();
    {  // convert staged fp32 tile -> transposed bf16 planes BT[c][k]
      int kk = tid >> 3, grp = tid & 7;
      const float2* src = &sWstage[kk * 64 + grp * 8];
#pragma unroll
      for (int i = 0; i < 8; ++i) {
        int c = grp * 8 + i;
        float2 v = src[i];
        sBre[c * BSTR + kk] = (__bf16)v.x;
        sBim[c * BSTR + kk] = (__bf16)v.y;
      }
    }
#else
    {  // direct stage of W tile 32x64 transposed -> BT[c][k]
      int kk = tid >> 3, grp = tid & 7;
      const float2* src = W + (size_t)(k0 + kk) * N + n0 + grp * 8;
#pragma unroll
      for (int i = 0; i < 8; ++i) {
        int c = grp * 8 + i;
        float2 v = src[i];
        sBre[c * BSTR + kk] = (__bf16)v.x;
        sBim[c * BSTR + kk] = (__bf16)v.y;
      }
    }
#endif
    __syncthreads();
    v16bf are[2], aim[2], bre[2], bim[2];
#pragma unroll
    for (int mt = 0; mt < 2; ++mt) {
      are[mt] = fragA(sAre, wm * 32 + mt * 16, lane);
      aim[mt] = fragA(sAim, wm * 32 + mt * 16, lane);
    }
#pragma unroll
    for (int nt = 0; nt < 2; ++nt) {
      bre[nt] = fragB(sBre, wn * 32 + nt * 16, lane);
      bim[nt] = fragB(sBim, wn * 32 + nt * 16, lane);
    }
#pragma unroll
    for (int mt = 0; mt < 2; ++mt)
#pragma unroll
      for (int nt = 0; nt < 2; ++nt) {
        accpp[mt][nt] = wmma_bf16(are[mt], bre[nt], accpp[mt][nt]);
        accmm[mt][nt] = wmma_bf16(aim[mt], bim[nt], accmm[mt][nt]);
        accim[mt][nt] = wmma_bf16(are[mt], bim[nt], accim[mt][nt]);
        accim[mt][nt] = wmma_bf16(aim[mt], bre[nt], accim[mt][nt]);
      }
    __syncthreads();
  }

  int cl = lane & 15, chs = (lane >> 4) * 8;
#pragma unroll
  for (int mt = 0; mt < 2; ++mt)
#pragma unroll
    for (int nt = 0; nt < 2; ++nt) {
      int tm0 = m0 + wm * 32 + mt * 16;
      int tn0 = n0 + wn * 32 + nt * 16;
      int colg = tn0 + cl;
      float2 bv = bias[colg];
#pragma unroll
      for (int r = 0; r < 8; ++r) {
        int mg = tm0 + r + chs;
        if (mg < M) {
          float re = accpp[mt][nt][r] - accmm[mt][nt][r] + bv.x;
          float im = accim[mt][nt][r] + bv.y;
          if (resid) {
            float2 rv = resid[(size_t)mg * N + colg];
            re += rv.x; im += rv.y;
          }
          C[(size_t)mg * N + colg] = make_float2(re, im);
        }
      }
    }
}

// ---------------------------------------------------------------------------
// Attention scores: S[bh,m,n] = 0.125 * Re(q . conj(k)) = 0.125*(qre.kre+qim.kim)
// ---------------------------------------------------------------------------
__global__ __launch_bounds__(256) void scores_k(const float2* __restrict__ qkv,
                                                float* __restrict__ S) {
  __shared__ __bf16 sQre[128 * ASTR];
  __shared__ __bf16 sQim[128 * ASTR];
  __shared__ __bf16 sKre[64 * BSTR];
  __shared__ __bf16 sKim[64 * BSTR];
  int tid = threadIdx.x, lane = tid & 31, w = tid >> 5;
  int wm = w >> 1, wn = w & 1;
  int bh = blockIdx.z, b = bh / NHEAD, hh = bh % NHEAD;
  int m0 = blockIdx.y * 128, n0 = blockIdx.x * 64;
  const v8f vz = {0.f, 0.f, 0.f, 0.f, 0.f, 0.f, 0.f, 0.f};
  v8f acc[2][2];
#pragma unroll
  for (int i = 0; i < 2; ++i)
#pragma unroll
    for (int j = 0; j < 2; ++j) acc[i][j] = vz;

  for (int k0 = 0; k0 < DHEAD; k0 += 32) {
    {  // stage Q tile
      int row = tid >> 1, half = tid & 1;
      int m = m0 + row;
      __bf16* dre = &sQre[row * ASTR + half * 16];
      __bf16* di  = &sQim[row * ASTR + half * 16];
      if (m < NTOK) {
        const float2* src = qkv + (size_t)(b * NTOK + m) * (3 * DMODEL) + hh * DHEAD + k0 + half * 16;
#pragma unroll
        for (int i = 0; i < 16; ++i) {
          float2 v = src[i];
          dre[i] = (__bf16)v.x; di[i] = (__bf16)v.y;
        }
      } else {
#pragma unroll
        for (int i = 0; i < 16; ++i) { dre[i] = (__bf16)0.f; di[i] = (__bf16)0.f; }
      }
    }
    {  // stage K^T: BT[c][kk] = K[n0+c][k0+kk]  (K rows are contiguous in d)
      int c = tid >> 2, q4 = tid & 3;
      int n = n0 + c;
      __bf16* dre = &sKre[c * BSTR + q4 * 8];
      __bf16* di  = &sKim[c * BSTR + q4 * 8];
      if (n < NTOK) {
        const float2* src = qkv + (size_t)(b * NTOK + n) * (3 * DMODEL) + DMODEL + hh * DHEAD + k0 + q4 * 8;
#pragma unroll
        for (int i = 0; i < 8; ++i) {
          float2 v = src[i];
          dre[i] = (__bf16)v.x; di[i] = (__bf16)v.y;
        }
      } else {
#pragma unroll
        for (int i = 0; i < 8; ++i) { dre[i] = (__bf16)0.f; di[i] = (__bf16)0.f; }
      }
    }
    __syncthreads();
    v16bf are[2], aim[2], bre[2], bim[2];
#pragma unroll
    for (int mt = 0; mt < 2; ++mt) {
      are[mt] = fragA(sQre, wm * 32 + mt * 16, lane);
      aim[mt] = fragA(sQim, wm * 32 + mt * 16, lane);
    }
#pragma unroll
    for (int nt = 0; nt < 2; ++nt) {
      bre[nt] = fragB(sKre, wn * 32 + nt * 16, lane);
      bim[nt] = fragB(sKim, wn * 32 + nt * 16, lane);
    }
#pragma unroll
    for (int mt = 0; mt < 2; ++mt)
#pragma unroll
      for (int nt = 0; nt < 2; ++nt) {
        acc[mt][nt] = wmma_bf16(are[mt], bre[nt], acc[mt][nt]);
        acc[mt][nt] = wmma_bf16(aim[mt], bim[nt], acc[mt][nt]);
      }
    __syncthreads();
  }

  int cl = lane & 15, chs = (lane >> 4) * 8;
#pragma unroll
  for (int mt = 0; mt < 2; ++mt)
#pragma unroll
    for (int nt = 0; nt < 2; ++nt) {
      int tm0 = m0 + wm * 32 + mt * 16;
      int tn0 = n0 + wn * 32 + nt * 16;
      int n = tn0 + cl;
      if (n < NTOK) {
#pragma unroll
        for (int r = 0; r < 8; ++r) {
          int m = tm0 + r + chs;
          if (m < NTOK) S[((size_t)bh * NPAD + m) * NPAD + n] = acc[mt][nt][r] * 0.125f;
        }
      }
    }
}

// ---------------------------------------------------------------------------
// Row softmax over 196 valid cols; writes zeros into padding cols [196,208)
// ---------------------------------------------------------------------------
__global__ __launch_bounds__(256) void softmax_k(float* __restrict__ S) {
  int wid = threadIdx.x >> 5, lane = threadIdx.x & 31;
  int m = blockIdx.x * 8 + wid;
  int bh = blockIdx.y;
  if (m >= NTOK) return;
  float* row = S + ((size_t)bh * NPAD + m) * NPAD;
  float vals[7];
  float mx = -3.4e38f;
#pragma unroll
  for (int i = 0; i < 7; ++i) {
    int c = lane + i * 32;
    float v = (c < NTOK) ? row[c] : -3.4e38f;
    vals[i] = v;
    mx = fmaxf(mx, v);
  }
#pragma unroll
  for (int o = 16; o > 0; o >>= 1) mx = fmaxf(mx, __shfl_xor(mx, o, 32));
  float sum = 0.f;
#pragma unroll
  for (int i = 0; i < 7; ++i) {
    int c = lane + i * 32;
    float e = (c < NTOK) ? __expf(vals[i] - mx) : 0.f;
    vals[i] = e;
    sum += e;
  }
#pragma unroll
  for (int o = 16; o > 0; o >>= 1) sum += __shfl_xor(sum, o, 32);
  float inv = 1.f / sum;
#pragma unroll
  for (int i = 0; i < 7; ++i) {
    int c = lane + i * 32;
    if (c < NPAD) row[c] = vals[i] * inv;
  }
}

// ---------------------------------------------------------------------------
// O[b, m, h*64+d] = sum_k attn[bh,m,k] * V[b,k,h,d]   (real x complex)
// ---------------------------------------------------------------------------
__global__ __launch_bounds__(256) void av_k(const float* __restrict__ S,
                                            const float2* __restrict__ qkv,
                                            float2* __restrict__ O) {
  __shared__ __bf16 sAt[128 * ASTR];
  __shared__ __bf16 sVre[64 * BSTR];
  __shared__ __bf16 sVim[64 * BSTR];
  int tid = threadIdx.x, lane = tid & 31, w = tid >> 5;
  int wm = w >> 1, wn = w & 1;
  int bh = blockIdx.z, b = bh / NHEAD, hh = bh % NHEAD;
  int m0 = blockIdx.y * 128;
  const v8f vz = {0.f, 0.f, 0.f, 0.f, 0.f, 0.f, 0.f, 0.f};
  v8f accre[2][2], accim[2][2];
#pragma unroll
  for (int i = 0; i < 2; ++i)
#pragma unroll
    for (int j = 0; j < 2; ++j) { accre[i][j] = vz; accim[i][j] = vz; }

  for (int k0 = 0; k0 < 224; k0 += 32) {
    {  // stage attn tile (real)
      int row = tid >> 1, half = tid & 1;
      int m = m0 + row;
      __bf16* d = &sAt[row * ASTR + half * 16];
#pragma unroll
      for (int i = 0; i < 16; ++i) {
        int k = k0 + half * 16 + i;
        float v = (m < NTOK && k < NPAD) ? S[((size_t)bh * NPAD + m) * NPAD + k] : 0.f;
        d[i] = (__bf16)v;
      }
    }
    {  // stage V^T: BT[d][kk] = V[k0+kk][d]
      int kk = tid >> 3, grp = tid & 7;
      int kg = k0 + kk;
      if (kg < NTOK) {
        const float2* src = qkv + (size_t)(b * NTOK + kg) * (3 * DMODEL) + 2 * DMODEL + hh * DHEAD + grp * 8;
#pragma unroll
        for (int i = 0; i < 8; ++i) {
          int d = grp * 8 + i;
          float2 v = src[i];
          sVre[d * BSTR + kk] = (__bf16)v.x;
          sVim[d * BSTR + kk] = (__bf16)v.y;
        }
      } else {
#pragma unroll
        for (int i = 0; i < 8; ++i) {
          int d = grp * 8 + i;
          sVre[d * BSTR + kk] = (__bf16)0.f;
          sVim[d * BSTR + kk] = (__bf16)0.f;
        }
      }
    }
    __syncthreads();
    v16bf at[2], bre[2], bim[2];
#pragma unroll
    for (int mt = 0; mt < 2; ++mt) at[mt] = fragA(sAt, wm * 32 + mt * 16, lane);
#pragma unroll
    for (int nt = 0; nt < 2; ++nt) {
      bre[nt] = fragB(sVre, wn * 32 + nt * 16, lane);
      bim[nt] = fragB(sVim, wn * 32 + nt * 16, lane);
    }
#pragma unroll
    for (int mt = 0; mt < 2; ++mt)
#pragma unroll
      for (int nt = 0; nt < 2; ++nt) {
        accre[mt][nt] = wmma_bf16(at[mt], bre[nt], accre[mt][nt]);
        accim[mt][nt] = wmma_bf16(at[mt], bim[nt], accim[mt][nt]);
      }
    __syncthreads();
  }

  int cl = lane & 15, chs = (lane >> 4) * 8;
#pragma unroll
  for (int mt = 0; mt < 2; ++mt)
#pragma unroll
    for (int nt = 0; nt < 2; ++nt) {
      int tm0 = m0 + wm * 32 + mt * 16;
      int col = hh * DHEAD + wn * 32 + nt * 16 + cl;
#pragma unroll
      for (int r = 0; r < 8; ++r) {
        int m = tm0 + r + chs;
        if (m < NTOK)
          O[(size_t)(b * NTOK + m) * DMODEL + col] = make_float2(accre[mt][nt][r], accim[mt][nt][r]);
      }
    }
}

// ---------------------------------------------------------------------------
// Complex whitening LayerNorm (torchcvnn): per-row joint (re,im) whitening
// ---------------------------------------------------------------------------
__device__ __forceinline__ float block_sum(float v, float* red) {
  int tid = threadIdx.x;
  __syncthreads();
  red[tid] = v;
  __syncthreads();
#pragma unroll
  for (int s = 128; s > 0; s >>= 1) {
    if (tid < s) red[tid] += red[tid + s];
    __syncthreads();
  }
  return red[0];
}

__global__ __launch_bounds__(256) void whiten_k(const float2* __restrict__ in,
                                                const float2* __restrict__ g,
                                                const float2* __restrict__ bb,
                                                float2* __restrict__ out, int C) {
  __shared__ float red[256];
  int r = blockIdx.x, tid = threadIdx.x;
  const float2* rowp = in + (size_t)r * C;
  float2 zc[12];
  int cnt = 0;
  float sre = 0.f, sim = 0.f;
  for (int c = tid, i = 0; c < C; c += 256, ++i) {
    float2 z = rowp[c];
    zc[i] = z;
    sre += z.x; sim += z.y;
    ++cnt;
  }
  sre = block_sum(sre, red);
  sim = block_sum(sim, red);
  float invC = 1.f / (float)C;
  float mre = sre * invC, mim = sim * invC;
  float vrr = 0.f, vii = 0.f, vri = 0.f;
  for (int i = 0; i < cnt; ++i) {
    float re = zc[i].x - mre, im = zc[i].y - mim;
    zc[i].x = re; zc[i].y = im;
    vrr += re * re; vii += im * im; vri += re * im;
  }
  vrr = block_sum(vrr, red);
  vii = block_sum(vii, red);
  vri = block_sum(vri, red);
  vrr = vrr * invC + EPSLN;
  vii = vii * invC + EPSLN;
  vri = vri * invC;
  float s  = sqrtf(vrr * vii - vri * vri);
  float t  = sqrtf(vrr + vii + 2.f * s);
  float st = s * t;
  float wrr = (vii + s) / st, wii = (vrr + s) / st, wri = -vri / st;
  float2* orow = out + (size_t)r * C;
  for (int c = tid, i = 0; c < C; c += 256, ++i) {
    float re = zc[i].x, im = zc[i].y;
    float ore = wrr * re + wri * im;
    float oim = wri * re + wii * im;
    float2 gv = g[c], bv = bb[c];
    orow[c] = make_float2(ore * gv.x - oim * gv.y + bv.x,
                          ore * gv.y + oim * gv.x + bv.y);
  }
}

// ---------------------------------------------------------------------------
// modReLU: z *= relu(|z| + b) / (|z| + 1e-6)
// ---------------------------------------------------------------------------
__global__ __launch_bounds__(256) void modrelu_k(float2* __restrict__ f,
                                                 const float* __restrict__ bptr, int n) {
  int i = blockIdx.x * 256 + threadIdx.x;
  if (i >= n) return;
  float b = *bptr;
  float2 z = f[i];
  float mag = sqrtf(z.x * z.x + z.y * z.y);
  float sc = fmaxf(mag + b, 0.f) / (mag + 1e-6f);
  f[i] = make_float2(sc * z.x, sc * z.y);
}

// ---------------------------------------------------------------------------
// Orchestration
// ---------------------------------------------------------------------------
extern "C" void kernel_launch(void* const* d_in, const int* in_sizes, int n_in,
                              void* d_out, int out_size, void* d_ws, size_t ws_size,
                              hipStream_t stream) {
  const float2* x       = (const float2*)d_in[0];
  const float2* patch_w = (const float2*)d_in[1];
  const float2* patch_b = (const float2*)d_in[2];
  const float2* ln1_g   = (const float2*)d_in[3];
  const float2* ln1_b   = (const float2*)d_in[4];
  const float2* Wqkv    = (const float2*)d_in[5];
  const float2* bqkv    = (const float2*)d_in[6];
  const float2* Wo      = (const float2*)d_in[7];
  const float2* bo      = (const float2*)d_in[8];
  const float2* ln2_g   = (const float2*)d_in[9];
  const float2* ln2_b   = (const float2*)d_in[10];
  const float2* W1      = (const float2*)d_in[11];
  const float2* b1      = (const float2*)d_in[12];
  const float2* lnm_g   = (const float2*)d_in[13];
  const float2* lnm_b   = (const float2*)d_in[14];
  const float*  mod_b   = (const float*)d_in[15];
  const float2* W2      = (const float2*)d_in[16];
  const float2* b2      = (const float2*)d_in[17];
  const float2* fin_g   = (const float2*)d_in[18];
  const float2* fin_b   = (const float2*)d_in[19];

  float* ws = (float*)d_ws;
  size_t off = 0;
  float2* Ap   = (float2*)(ws + off); off += (size_t)MROWS * DMODEL * 2;
  float2* hbuf = (float2*)(ws + off); off += (size_t)MROWS * DMODEL * 2;
  float2* nbuf = (float2*)(ws + off); off += (size_t)MROWS * DMODEL * 2;
  float2* qkvb = (float2*)(ws + off); off += (size_t)MROWS * 3 * DMODEL * 2;
  float*  Sbuf = (ws + off);          off += (size_t)BSZ * NHEAD * NPAD * NPAD;
  float2* obuf = (float2*)(ws + off); off += (size_t)MROWS * DMODEL * 2;
  float2* fbuf = (float2*)(ws + off); off += (size_t)MROWS * FDIM * 2;

  dim3 blk(256);
  const int GM = (MROWS + 127) / 128;  // 25

  pack_patches_k<<<(MROWS * DMODEL + 255) / 256, blk, 0, stream>>>(x, Ap);
  cgemm_cc_k<<<dim3(DMODEL / 64, GM), blk, 0, stream>>>(Ap, patch_w, patch_b, nullptr,
                                                        hbuf, MROWS, DMODEL, DMODEL);

  for (int l = 0; l < NLAYER; ++l) {
    whiten_k<<<MROWS, blk, 0, stream>>>(hbuf, ln1_g + (size_t)l * DMODEL,
                                        ln1_b + (size_t)l * DMODEL, nbuf, DMODEL);
    cgemm_cc_k<<<dim3(3 * DMODEL / 64, GM), blk, 0, stream>>>(
        nbuf, Wqkv + (size_t)l * DMODEL * 3 * DMODEL, bqkv + (size_t)l * 3 * DMODEL,
        nullptr, qkvb, MROWS, 3 * DMODEL, DMODEL);
    scores_k<<<dim3(4, 2, BSZ * NHEAD), blk, 0, stream>>>(qkvb, Sbuf);
    softmax_k<<<dim3((NTOK + 7) / 8, BSZ * NHEAD), blk, 0, stream>>>(Sbuf);
    av_k<<<dim3(1, 2, BSZ * NHEAD), blk, 0, stream>>>(Sbuf, qkvb, obuf);
    cgemm_cc_k<<<dim3(DMODEL / 64, GM), blk, 0, stream>>>(
        obuf, Wo + (size_t)l * DMODEL * DMODEL, bo + (size_t)l * DMODEL,
        hbuf, hbuf, MROWS, DMODEL, DMODEL);
    whiten_k<<<MROWS, blk, 0, stream>>>(hbuf, ln2_g + (size_t)l * DMODEL,
                                        ln2_b + (size_t)l * DMODEL, nbuf, DMODEL);
    cgemm_cc_k<<<dim3(FDIM / 64, GM), blk, 0, stream>>>(
        nbuf, W1 + (size_t)l * DMODEL * FDIM, b1 + (size_t)l * FDIM,
        nullptr, fbuf, MROWS, FDIM, DMODEL);
    whiten_k<<<MROWS, blk, 0, stream>>>(fbuf, lnm_g + (size_t)l * FDIM,
                                        lnm_b + (size_t)l * FDIM, fbuf, FDIM);
    modrelu_k<<<(MROWS * FDIM + 255) / 256, blk, 0, stream>>>(fbuf, mod_b + l,
                                                              MROWS * FDIM);
    cgemm_cc_k<<<dim3(DMODEL / 64, GM), blk, 0, stream>>>(
        fbuf, W2 + (size_t)l * FDIM * DMODEL, b2 + (size_t)l * DMODEL,
        hbuf, hbuf, MROWS, DMODEL, FDIM);
  }
  whiten_k<<<MROWS, blk, 0, stream>>>(hbuf, fin_g, fin_b, (float2*)d_out, DMODEL);

  (void)in_sizes; (void)n_in; (void)out_size; (void)ws_size;
}
